// BlockDrop_24455543783992
// MI455X (gfx1250) — matrix operand order repaired and checked by reading, compiled, not run on hardware
//
#include <hip/hip_runtime.h>
#include <hip/hip_bf16.h>

typedef _Float16 f16;
typedef __attribute__((ext_vector_type(16))) _Float16 v16h;
typedef __attribute__((ext_vector_type(8)))  float    v8f;
typedef __attribute__((ext_vector_type(4)))  unsigned int u32x4;
typedef __attribute__((ext_vector_type(4)))  int i32x4;
typedef __attribute__((ext_vector_type(8)))  int i32x8;

struct __align__(16) U128 { unsigned int x, y, z, w; };
union Frag { U128 u[2]; v16h v; };

// ---------------------------------------------------------------------------
// TDM: 2D tile load Global -> LDS. Tile = 32 (k, contiguous) x 64 (co rows),
// row pitch = pitchElems (f16 elements). D# per CDNA5 ISA.
// ---------------------------------------------------------------------------
__device__ __forceinline__ void tdm_load_tile_64x32(f16* ldsDst, const f16* gsrc,
                                                    int pitchElems) {
  unsigned long long ga = (unsigned long long)(const void*)gsrc;
  u32x4 g0;
  g0[0] = 1u;                                             // count=1, user mode
  g0[1] = (unsigned)(unsigned long long)(void*)ldsDst;    // lds_addr [63:32]
  g0[2] = (unsigned)ga;                                   // global_addr lo
  g0[3] = ((unsigned)(ga >> 32) & 0x01FFFFFFu)            // global_addr [56:32]
          | 0x80000000u;                                  // type=2 (image)
  i32x8 g1;
  g1[0] = 1 << 16;          // workgroup_mask=0, data_size=1 (2 bytes)
  g1[1] = 32 << 16;         // tensor_dim0 = 32 (x extent)
  g1[2] = 64 << 16;         // tensor_dim0 hi=0 | tensor_dim1 = 64
  g1[3] = 32 << 16;         // tensor_dim1 hi=0 | tile_dim0 = 32
  g1[4] = 64;               // tile_dim1 = 64, tile_dim2 = 0
  g1[5] = pitchElems;       // tensor_dim0_stride (lo32) = row pitch
  g1[6] = 0;
  g1[7] = 0;
  i32x4 z4 = {0, 0, 0, 0};
#if __clang_major__ >= 23
  i32x8 z8 = {0, 0, 0, 0, 0, 0, 0, 0};
  __builtin_amdgcn_tensor_load_to_lds(g0, g1, z4, z4, z8, 0);
#else
  __builtin_amdgcn_tensor_load_to_lds(g0, g1, z4, z4, 0);
#endif
}

// ---------------------------------------------------------------------------
// Implicit-GEMM conv, NHWC f16 activations, weights packed [Co][Kpad] f16 with
// k = (ky*ks+kx)*Ci + ci. Block = 256 threads = 8 waves; block tile = 128 px x
// 64 cout; each wave owns 16 px x 64 cout -> 4 v_wmma_f32_16x16x32_f16 per
// K-step sharing one A fragment. Triple-buffered LDS, one barrier per K-step;
// wave 0 keeps <=2 TDM weight-tile loads in flight (s_wait_tensorcnt 1).
// ---------------------------------------------------------------------------
__global__ __launch_bounds__(256) void conv_wmma_kernel(
    const f16* __restrict__ x, const f16* __restrict__ wT,
    const float* __restrict__ bias, f16* __restrict__ y,
    int Hi, int Wi, int Ci, int Co, int Ho, int Wo,
    int ks, int stride, int pad, int Kpad, int relu, int vecA) {
  __shared__ __align__(16) f16 Atile[3][128 * 32];   // [pixel][k]
  __shared__ __align__(16) f16 Btile[3][64 * 32];    // [co][k] (TDM-written)

  const int tid = threadIdx.x;
  const int lane = tid & 31;
  const int wid = tid >> 5;                 // 8 waves, 16 pixels each
  const long long pBase = (long long)blockIdx.x * 128;
  const int coBase = blockIdx.y * 64;
  const int HoWo = Ho * Wo;
  const int K = ks * ks * Ci;
  const int nSteps = Kpad / 32;

  // pixel this thread stages for the A tile (2 threads per pixel row)
  const int aRow = tid >> 1;                // 0..127
  const int aK16 = (tid & 1) * 16;          // 16 k-elements per thread
  const long long pA = pBase + aRow;
  const int nIdx = (int)(pA / HoWo);
  const int prem = (int)(pA - (long long)nIdx * HoWo);
  const int ho = prem / Wo;
  const int wo = prem - ho * Wo;

  auto stageA = [&](int step, int buf) {
    const int kk = step * 32;
    f16* dst = &Atile[buf][aRow * 32 + aK16];
    if (vecA) {  // Ci % 32 == 0: aligned 16-run of k is contiguous ci
      const int k = kk + aK16;
      const int r = k / Ci;
      const int cb = k - r * Ci;
      const int ky = r / ks;
      const int kx = r - ky * ks;
      const int hi = ho * stride - pad + ky;
      const int wi = wo * stride - pad + kx;
      U128 v0, v1;
      v0.x = v0.y = v0.z = v0.w = 0u;
      v1 = v0;
      if (hi >= 0 && hi < Hi && wi >= 0 && wi < Wi) {
        const U128* s =
            (const U128*)(x + (((long long)nIdx * Hi + hi) * Wi + wi) * Ci + cb);
        v0 = s[0];
        v1 = s[1];
      }
      ((U128*)dst)[0] = v0;
      ((U128*)dst)[1] = v1;
    } else {     // seed conv (Ci=3, K=27, zero-padded to 32)
      for (int j = 0; j < 16; ++j) {
        const int k = kk + aK16 + j;
        f16 v = (f16)0.f;
        if (k < K) {
          const int r = k / Ci;
          const int ci = k - r * Ci;
          const int ky = r / ks;
          const int kx = r - ky * ks;
          const int hi = ho * stride - pad + ky;
          const int wi = wo * stride - pad + kx;
          if (hi >= 0 && hi < Hi && wi >= 0 && wi < Wi)
            v = x[(((long long)nIdx * Hi + hi) * Wi + wi) * Ci + ci];
        }
        dst[j] = v;
      }
    }
  };

  v8f acc0 = {}, acc1 = {}, acc2 = {}, acc3 = {};

  // prologue: fill pipeline stage 0
  stageA(0, 0);
  if (tid < 32) tdm_load_tile_64x32(&Btile[0][0], wT + (long long)coBase * Kpad,
                                    Kpad);

  for (int i = 0; i < nSteps; ++i) {
    const int cb = i % 3;
    const int nb = (i + 1) % 3;
    if (i + 1 < nSteps) {
      stageA(i + 1, nb);
      if (tid < 32)
        tdm_load_tile_64x32(&Btile[nb][0],
                            wT + (long long)coBase * Kpad + (i + 1) * 32, Kpad);
    }
    if (tid < 32) {
      if (i + 1 < nSteps)
        __builtin_amdgcn_s_wait_tensorcnt(1);  // step i's tile resident
      else
        __builtin_amdgcn_s_wait_tensorcnt(0);
    }
    __syncthreads();

    const f16* As = &Atile[cb][0];
    const f16* Bs = &Btile[cb][0];
    const int mRow = wid * 16 + (lane & 15);
    const int c0 = (lane >> 4) * 8;          // A: k-runs [c0,+8) & [c0+16,+8)
    Frag a;
    a.u[0] = *(const U128*)(As + mRow * 32 + c0);
    a.u[1] = *(const U128*)(As + mRow * 32 + c0 + 16);
    const int kb = (lane >> 4) * 8;          // B: k-runs [kb,+8) & [kb+16,+8)
    const int cn = lane & 15;
    Frag b0, b1, b2, b3;
    b0.u[0] = *(const U128*)(Bs + (cn +  0) * 32 + kb);
    b0.u[1] = *(const U128*)(Bs + (cn +  0) * 32 + kb + 16);
    b1.u[0] = *(const U128*)(Bs + (cn + 16) * 32 + kb);
    b1.u[1] = *(const U128*)(Bs + (cn + 16) * 32 + kb + 16);
    b2.u[0] = *(const U128*)(Bs + (cn + 32) * 32 + kb);
    b2.u[1] = *(const U128*)(Bs + (cn + 32) * 32 + kb + 16);
    b3.u[0] = *(const U128*)(Bs + (cn + 48) * 32 + kb);
    b3.u[1] = *(const U128*)(Bs + (cn + 48) * 32 + kb + 16);
    acc0 = __builtin_amdgcn_wmma_f32_16x16x32_f16(false, a.v, false, b0.v,
                                                  (short)0, acc0, false, false);
    acc1 = __builtin_amdgcn_wmma_f32_16x16x32_f16(false, a.v, false, b1.v,
                                                  (short)0, acc1, false, false);
    acc2 = __builtin_amdgcn_wmma_f32_16x16x32_f16(false, a.v, false, b2.v,
                                                  (short)0, acc2, false, false);
    acc3 = __builtin_amdgcn_wmma_f32_16x16x32_f16(false, a.v, false, b3.v,
                                                  (short)0, acc3, false, false);
  }

  // ---- epilogue: bias (+ReLU), f32 -> f16 store, D layout M=v+8*(l>=16) ----
  const int cn = lane & 15;
  const int co0 = coBase + cn;
  const float bv0 = bias[co0];
  const float bv1 = bias[co0 + 16];
  const float bv2 = bias[co0 + 32];
  const float bv3 = bias[co0 + 48];
  const int hilo = (lane >> 4) * 8;
  for (int e = 0; e < 8; ++e) {
    const long long pOut = pBase + wid * 16 + hilo + e;
    float f0 = acc0[e] + bv0;
    float f1 = acc1[e] + bv1;
    float f2 = acc2[e] + bv2;
    float f3 = acc3[e] + bv3;
    if (relu) {
      f0 = fmaxf(f0, 0.f); f1 = fmaxf(f1, 0.f);
      f2 = fmaxf(f2, 0.f); f3 = fmaxf(f3, 0.f);
    }
    f16* yp = y + pOut * Co + co0;
    yp[0]  = (f16)f0;
    yp[16] = (f16)f1;
    yp[32] = (f16)f2;
    yp[48] = (f16)f3;
  }
}

// ---------------------------------------------------------------------------
// Policy-gated residual combine: out = m*relu(res+fx) + (1-m)*res, m per batch
// ---------------------------------------------------------------------------
__global__ void combine_kernel(const f16* __restrict__ res,
                               const f16* __restrict__ fx,
                               const float* __restrict__ probs, int pidx,
                               f16* __restrict__ out, long long total, int Co,
                               int HoWo) {
  const long long i8 = ((long long)blockIdx.x * blockDim.x + threadIdx.x) * 8;
  if (i8 >= total) return;
  const long long p = i8 / Co;
  const int n = (int)(p / HoWo);
  const float m = (probs[n * 15 + pidx] < 0.5f) ? 0.f : 1.f;
  U128 rv = *(const U128*)(res + i8);
  U128 fv = *(const U128*)(fx + i8);
  const f16* rp = (const f16*)&rv;
  const f16* fp = (const f16*)&fv;
  U128 ov;
  f16* op = (f16*)&ov;
  for (int j = 0; j < 8; ++j) {
    const float r = (float)rp[j];
    const float s = fmaxf(r + (float)fp[j], 0.f);
    op[j] = (f16)(m * s + (1.f - m) * r);
  }
  *(U128*)(out + i8) = ov;
}

// Global average pool: [128,16,16,256] NHWC -> pooled f16 [128][256]
__global__ void pool_kernel(const f16* __restrict__ x, f16* __restrict__ pooled) {
  const int n = blockIdx.x;
  const int c = threadIdx.x;  // 256 threads = 256 channels
  float s = 0.f;
  const f16* base = x + (long long)n * 256 * 256 + c;
  for (int p = 0; p < 256; ++p) s += (float)base[p * 256];
  pooled[n * 256 + c] = (f16)(s * (1.f / 256.f));
}

// FC via WMMA: M=128 batch, N=1008 (1000 padded), K=256; one wave per 16x16
__global__ __launch_bounds__(256) void fc_wmma_kernel(
    const f16* __restrict__ pooled, const f16* __restrict__ fcT,
    const float* __restrict__ fcb, float* __restrict__ out) {
  const int lane = threadIdx.x & 31;
  const int gw = blockIdx.x * 8 + (threadIdx.x >> 5);  // 63 blocks * 8 = 504
  const int mT = gw / 63;
  const int nT = gw - mT * 63;
  const int row = mT * 16 + (lane & 15);
  const int c0 = (lane >> 4) * 8;
  const int co = nT * 16 + (lane & 15);
  const int kb = (lane >> 4) * 8;
  v8f acc = {};
  for (int kk = 0; kk < 256; kk += 32) {
    Frag a, b;
    a.u[0] = *(const U128*)(pooled + row * 256 + kk + c0);
    a.u[1] = *(const U128*)(pooled + row * 256 + kk + c0 + 16);
    b.u[0] = *(const U128*)(fcT + (long long)co * 256 + kk + kb);
    b.u[1] = *(const U128*)(fcT + (long long)co * 256 + kk + kb + 16);
    acc = __builtin_amdgcn_wmma_f32_16x16x32_f16(false, a.v, false, b.v,
                                                 (short)0, acc, false, false);
  }
  const int hilo = (lane >> 4) * 8;
  if (co < 1000) {
    const float bv = fcb[co];
    for (int e = 0; e < 8; ++e)
      out[(mT * 16 + hilo + e) * 1000 + co] = acc[e] + bv;
  }
}

// Input f32 NCHW -> f16 NHWC
__global__ void nchw2nhwc_kernel(const float* __restrict__ in,
                                 f16* __restrict__ out, int N, int C, int H,
                                 int W) {
  const long long i = (long long)blockIdx.x * blockDim.x + threadIdx.x;
  const long long total = (long long)N * C * H * W;
  if (i >= total) return;
  const int c = (int)(i % C);
  long long t = i / C;
  const int w = (int)(t % W); t /= W;
  const int h = (int)(t % H);
  const int n = (int)(t / H);
  out[i] = (f16)in[(((long long)n * C + c) * H + h) * W + w];
}

// Weights f32 [CoSrc][Ci][ks][ks] -> f16 [CoOut][Kpad], k=(ky*ks+kx)*Ci+ci
__global__ void pack_w_kernel(const float* __restrict__ w, f16* __restrict__ wT,
                              int CoOut, int CoSrc, int Ci, int ks, int Kpad) {
  const long long i = (long long)blockIdx.x * blockDim.x + threadIdx.x;
  if (i >= (long long)CoOut * Kpad) return;
  const int k = (int)(i % Kpad);
  const int co = (int)(i / Kpad);
  const int K = ks * ks * Ci;
  f16 v = (f16)0.f;
  if (co < CoSrc && k < K) {
    const int r = k / Ci;
    const int ci = k - r * Ci;
    const int ky = r / ks;
    const int kx = r - ky * ks;
    v = (f16)w[(((long long)co * Ci + ci) * ks + ky) * ks + kx];
  }
  wT[i] = v;
}

// ---------------------------------------------------------------------------
extern "C" void kernel_launch(void* const* d_in, const int* in_sizes, int n_in,
                              void* d_out, int out_size, void* d_ws,
                              size_t ws_size, hipStream_t stream) {
  (void)in_sizes; (void)n_in; (void)out_size; (void)ws_size;
  const float* inputs = (const float*)d_in[0];
  const float* probs = (const float*)d_in[1];
  const float* seed_w = (const float*)d_in[2];
  const float* seed_b = (const float*)d_in[3];
  const float* fc_w = (const float*)d_in[34];
  const float* fc_b = (const float*)d_in[35];

  char* ws = (char*)d_ws;
  size_t off = 0;
  auto alloc = [&](size_t bytes) -> void* {
    off = (off + 255) & ~(size_t)255;
    void* p = ws + off;
    off += bytes;
    return p;
  };

  const size_t ACT = (size_t)128 * 64 * 64 * 64 * 2;  // 64 MiB, max activation
  f16* act[3];
  for (int i = 0; i < 3; ++i) act[i] = (f16*)alloc(ACT);
  f16* pooled = (f16*)alloc(128 * 256 * 2);
  f16* seedT = (f16*)alloc(64 * 32 * 2);
  f16* fcT = (f16*)alloc((size_t)1008 * 256 * 2);

  const int cins[3] = {64, 64, 128};
  const int couts[3] = {64, 128, 256};
  const int strs[3] = {1, 2, 2};

  struct GW { f16 *ds, *w1, *w2, *rw1[4], *rw2[4]; } gw[3];
  for (int g = 0; g < 3; ++g) {
    const int ci = cins[g], co = couts[g];
    gw[g].ds = (f16*)alloc((size_t)co * ci * 2);
    gw[g].w1 = (f16*)alloc((size_t)co * ci * 9 * 2);
    gw[g].w2 = (f16*)alloc((size_t)co * co * 9 * 2);
    for (int j = 0; j < 4; ++j) {
      gw[g].rw1[j] = (f16*)alloc((size_t)co * co * 9 * 2);
      gw[g].rw2[j] = (f16*)alloc((size_t)co * co * 9 * 2);
    }
  }

  auto pack = [&](const float* src, f16* dst, int CoOut, int CoSrc, int Ci,
                  int ks, int Kpad) {
    const long long total = (long long)CoOut * Kpad;
    pack_w_kernel<<<dim3((unsigned)((total + 255) / 256)), dim3(256), 0,
                    stream>>>(src, dst, CoOut, CoSrc, Ci, ks, Kpad);
  };

  pack(seed_w, seedT, 64, 64, 3, 3, 32);  // K=27 -> Kpad=32
  for (int g = 0; g < 3; ++g) {
    const int ci = cins[g], co = couts[g];
    pack((const float*)d_in[4 + g * 10 + 0], gw[g].ds, co, co, ci, 1, ci);
    pack((const float*)d_in[4 + g * 10 + 2], gw[g].w1, co, co, ci, 3, ci * 9);
    pack((const float*)d_in[4 + g * 10 + 4], gw[g].w2, co, co, co, 3, co * 9);
    const float* rw1 = (const float*)d_in[4 + g * 10 + 6];
    const float* rw2 = (const float*)d_in[4 + g * 10 + 8];
    for (int j = 0; j < 4; ++j) {
      pack(rw1 + (long long)j * co * co * 9, gw[g].rw1[j], co, co, co, 3, co * 9);
      pack(rw2 + (long long)j * co * co * 9, gw[g].rw2[j], co, co, co, 3, co * 9);
    }
  }
  pack(fc_w, fcT, 1008, 1000, 256, 1, 256);

  {
    const long long total = (long long)128 * 3 * 64 * 64;
    nchw2nhwc_kernel<<<dim3((unsigned)((total + 255) / 256)), dim3(256), 0,
                       stream>>>(inputs, act[0], 128, 3, 64, 64);
  }

  auto conv = [&](const f16* xin, const f16* wTp, const float* b, f16* yout,
                  int Hi, int Ci, int Co, int ks, int stride, int relu) {
    const int pad = ks / 2;
    const int Ho = (Hi + 2 * pad - ks) / stride + 1;
    const int K = ks * ks * Ci;
    const int Kpad = (K + 31) & ~31;
    const int P = 128 * Ho * Ho;
    const int vecA = (Ci % 32) == 0;
    conv_wmma_kernel<<<dim3((unsigned)(P / 128), (unsigned)(Co / 64)),
                       dim3(256), 0, stream>>>(xin, wTp, b, yout, Hi, Hi, Ci,
                                               Co, Ho, Ho, ks, stride, pad,
                                               Kpad, relu, vecA);
  };

  auto combine = [&](const f16* res, const f16* fx, int pidx, f16* outp, int Ho,
                     int Co) {
    const long long total = (long long)128 * Ho * Ho * Co;
    combine_kernel<<<dim3((unsigned)(total / 8 / 256)), dim3(256), 0, stream>>>(
        res, fx, probs, pidx, outp, total, Co, Ho * Ho);
  };

  // seed conv + relu
  conv(act[0], seedT, seed_b, act[1], 64, 3, 64, 3, 1, 1);

  f16* cur = act[1];
  f16* t0 = act[0];
  f16* t1 = act[2];
  int Hc = 64, Cc = 64, pidx = 0;
  for (int g = 0; g < 3; ++g) {
    const int co = couts[g], s = strs[g];
    const float* ds_b = (const float*)d_in[4 + g * 10 + 1];
    const float* b1 = (const float*)d_in[4 + g * 10 + 3];
    const float* b2 = (const float*)d_in[4 + g * 10 + 5];
    const float* rb1 = (const float*)d_in[4 + g * 10 + 7];
    const float* rb2 = (const float*)d_in[4 + g * 10 + 9];
    const int Ho = (s == 2) ? Hc / 2 : Hc;

    // block 0: strided, 1x1 downsample residual
    conv(cur, gw[g].ds, ds_b, t0, Hc, Cc, co, 1, s, 0);   // R   -> t0
    conv(cur, gw[g].w1, b1, t1, Hc, Cc, co, 3, s, 1);     // H   -> t1
    conv(t1, gw[g].w2, b2, cur, Ho, co, co, 3, 1, 0);     // F   -> cur (x dead)
    combine(t0, cur, pidx++, t1, Ho, co);                 // x'  -> t1
    { f16* nc = t1; t1 = cur; cur = nc; }
    Hc = Ho; Cc = co;

    // 4 identity-residual blocks
    for (int j = 0; j < 4; ++j) {
      conv(cur, gw[g].rw1[j], rb1 + j * co, t0, Hc, co, co, 3, 1, 1);  // H
      conv(t0, gw[g].rw2[j], rb2 + j * co, t1, Hc, co, co, 3, 1, 0);   // F
      combine(cur, t1, pidx++, t0, Hc, co);                            // x'
      { f16* nc = t0; t0 = cur; cur = nc; }
    }
  }

  pool_kernel<<<dim3(128), dim3(256), 0, stream>>>(cur, pooled);
  fc_wmma_kernel<<<dim3(63), dim3(256), 0, stream>>>(pooled, fcT, fc_b,
                                                     (float*)d_out);
}
